// DisentangledMaskAttention_78082505441288
// MI455X (gfx1250) — matrix operand order repaired and checked by reading, not compiled, not run on hardware
//
#include <hip/hip_runtime.h>
#include <hip/hip_bf16.h>
#include <stddef.h>

// ---------------------------------------------------------------------------
// DisentangledMaskAttention for MI455X (gfx1250), wave32 + WMMA bf16/f32 +
// Tensor Data Mover (async tensor loads into LDS).
// B=8, L=512, D=1024, H=16, Dk=64, C=4.  Workspace: ~71 MB.
// ---------------------------------------------------------------------------

typedef __attribute__((ext_vector_type(16))) __bf16 v16bf;
typedef __attribute__((ext_vector_type(8)))  float  v8f;
typedef __attribute__((ext_vector_type(2)))  float  v2f;
typedef __attribute__((ext_vector_type(4)))  unsigned int u32x4;
typedef __attribute__((ext_vector_type(8)))  int    i32x8;
typedef __attribute__((ext_vector_type(4)))  int    i32x4;

#define ENABLE_WMMA_X4 1   // v_wmma_f32_16x16x4_f32 for cp.cp^T Gram tiles
#define ENABLE_TDM     1   // tensor_load_to_lds for attention K/V tiles

#define NB  8
#define NL  512
#define ND  1024
#define NH  16
#define NDK 64
#define NC  4
#define PI_F 3.14159265358979f

// ======================= TDM 2D tile load to LDS ===========================
// Builds a Tensor-DMA descriptor (ISA 8.3/8.4) for a 2D fp32 tile:
//   tile_w contiguous elems per row, tile_h rows, row stride = stride_elems,
//   LDS row pitch = tile_w + 1 dwords (pad_interval code 5 = 64 dwords,
//   pad_amount code 0 = 1 dword)  -> matches float KV[64][65].
__device__ __forceinline__ void dm_tdm_load_2d(unsigned lds_addr,
                                               const float* gptr,
                                               unsigned tile_w, unsigned tile_h,
                                               unsigned stride_elems) {
#if ENABLE_TDM
  unsigned long long ga = (unsigned long long)gptr;
  u32x4 g0;
  g0[0] = 1u;                                         // count=1, user mode
  g0[1] = lds_addr;                                   // LDS byte address
  g0[2] = (unsigned)(ga & 0xFFFFFFFFu);               // global_addr[31:0]
  g0[3] = (unsigned)((ga >> 32) & 0x01FFFFFFu) | (2u << 30);  // ga[56:32]|type=2
  i32x8 g1;
  g1[0] = (int)((2u << 16) | (1u << 20) | (5u << 22));// data_size=4B, pad_en,
                                                      // pad_interval=64dw(+1dw)
  g1[1] = (int)((tile_w & 0xFFFFu) << 16);            // tensor_dim0 lo16
  g1[2] = (int)(((tile_w >> 16) & 0xFFFFu) |          // tensor_dim0 hi16
                ((tile_h & 0xFFFFu) << 16));          // tensor_dim1 lo16
  g1[3] = (int)(((tile_h >> 16) & 0xFFFFu) |          // tensor_dim1 hi16
                ((tile_w & 0xFFFFu) << 16));          // tile_dim0
  g1[4] = (int)(tile_h & 0xFFFFu);                    // tile_dim1 (tile_dim2=0)
  g1[5] = (int)stride_elems;                          // tensor_dim0_stride lo32
  g1[6] = 0;                                          // stride hi / dim1_stride
  g1[7] = 0;
  i32x4 z4 = {0, 0, 0, 0};
#if defined(__clang_major__) && __clang_major__ >= 23
  i32x8 z8 = {0, 0, 0, 0, 0, 0, 0, 0};
  __builtin_amdgcn_tensor_load_to_lds(g0, g1, z4, z4, z8, 0);
#else
  __builtin_amdgcn_tensor_load_to_lds(g0, g1, z4, z4, 0);
#endif
#endif
}

// ============================== zero helper ================================
__global__ void dm_zero_kernel(float* __restrict__ p, int n) {
  int i = blockIdx.x * 256 + threadIdx.x;
  if (i < n) p[i] = 0.0f;
}

// ====================== bf16 WMMA GEMM: C = A*W + bias =====================
// A: MxK fp32 row-major, W: KxN fp32 row-major, C: MxN fp32.
// Block tile 128x64, K-stage 64; 8 waves; each wave owns 32x32 output
// (2x2 WMMA tiles) -> 8 v_wmma per stage with 2x2 fragment reuse.
__global__ __launch_bounds__(256) void dm_gemm_bias_kernel(
    const float* __restrict__ A, const float* __restrict__ W,
    const float* __restrict__ bias, float* __restrict__ Cout,
    int M, int N, int K)
{
  __shared__ __bf16 As[128][64 + 4];
  __shared__ __bf16 Bs[64][64 + 4];
  const int tid = threadIdx.x;
  const int lane = tid & 31, wv = tid >> 5;
  const int wr = wv & 3, wc = wv >> 2;      // 4 M-strips x 2 N-strips
  const int r = lane & 15, hlf = lane >> 4;
  const int m0 = blockIdx.x * 128;
  const int n0 = blockIdx.y * 64;

  v8f acc[2][2] = {};
  for (int k0 = 0; k0 < K; k0 += 64) {
    for (int i = tid; i < 128 * 64; i += 256) {
      int mm = i >> 6, kk = i & 63;
      As[mm][kk] = (__bf16)A[(size_t)(m0 + mm) * K + k0 + kk];
    }
    for (int i = tid; i < 64 * 64; i += 256) {
      int kk = i >> 6, nn = i & 63;
      Bs[kk][nn] = (__bf16)W[(size_t)(k0 + kk) * N + n0 + nn];
    }
    if (k0 + 64 < K) {  // prefetch next tiles into L2 (global_prefetch_b8)
      __builtin_prefetch(&A[(size_t)(m0 + (tid & 127)) * K + k0 + 64], 0, 0);
      __builtin_prefetch(&W[(size_t)(k0 + 64 + (tid & 63)) * N + n0], 0, 0);
    }
    __syncthreads();
#pragma unroll
    for (int ks = 0; ks < 64; ks += 32) {
      v16bf af0, af1, bf0, bf1;
#pragma unroll
      for (int j = 0; j < 16; ++j) {
        int km = ks + ((j >> 3) << 4) + (hlf << 3) + (j & 7);  // A K map
        af0[j] = As[wr * 32 + r][km];
        af1[j] = As[wr * 32 + 16 + r][km];
        bf0[j] = Bs[km][wc * 32 + r];
        bf1[j] = Bs[km][wc * 32 + 16 + r];
      }
      acc[0][0] = __builtin_amdgcn_wmma_f32_16x16x32_bf16(false, af0, false, bf0,
                                                          (short)0, acc[0][0], false, false);
      acc[0][1] = __builtin_amdgcn_wmma_f32_16x16x32_bf16(false, af0, false, bf1,
                                                          (short)0, acc[0][1], false, false);
      acc[1][0] = __builtin_amdgcn_wmma_f32_16x16x32_bf16(false, af1, false, bf0,
                                                          (short)0, acc[1][0], false, false);
      acc[1][1] = __builtin_amdgcn_wmma_f32_16x16x32_bf16(false, af1, false, bf1,
                                                          (short)0, acc[1][1], false, false);
    }
    __syncthreads();
  }
#pragma unroll
  for (int t = 0; t < 2; ++t)
#pragma unroll
    for (int u = 0; u < 2; ++u)
#pragma unroll
      for (int i = 0; i < 8; ++i) {
        int mm = m0 + wr * 32 + t * 16 + hlf * 8 + i;
        int nn = n0 + wc * 32 + u * 16 + r;
        Cout[(size_t)mm * N + nn] = acc[t][u][i] + bias[nn];
      }
}

// ================= projection to H=16 outputs, transposed ==================
__global__ __launch_bounds__(64) void dm_proj16_kernel(
    const float* __restrict__ X, const float* __restrict__ W16,
    const float* __restrict__ b16, float* __restrict__ out)
{
  int row = blockIdx.x;                 // b*L + l
  int tid = threadIdx.x;
  int h = tid & 15, chunk = tid >> 4;
  const float* x = X + (size_t)row * ND;
  float s = 0.f;
  int k0 = chunk * (ND / 4);
  for (int kk = k0; kk < k0 + ND / 4; ++kk)
    s += x[kk] * W16[(size_t)kk * NH + h];
  __shared__ float red[64];
  red[tid] = s;
  __syncthreads();
  if (tid < NH) {
    float v = red[h] + red[16 + h] + red[32 + h] + red[48 + h] + b16[h];
    int b = row / NL, l = row % NL;
    out[((size_t)b * NH + h) * NL + l] = v;
  }
}

// ============================ clustering ===================================
__global__ __launch_bounds__(256) void dm_clustering_kernel(
    const float* __restrict__ Z, const float* __restrict__ alv,
    const float* __restrict__ mu, const float* __restrict__ log_var,
    const float* __restrict__ log_prior,
    float* __restrict__ cp, float* __restrict__ lpdf_out,
    float* __restrict__ kl_out, unsigned int* __restrict__ gmax_enc,
    int muHStride, int doMax)
{
  int idx = blockIdx.x * 256 + threadIdx.x;
  if (idx >= NB * NH * NL) return;
  int l = idx & (NL - 1);
  int h = (idx >> 9) & (NH - 1);
  int b = idx >> 13;

  const float* z = Z + ((size_t)b * NL + l) * ND + h * NDK;
  int hm = muHStride ? h : 0;
  const float* mup = mu + (size_t)hm * NC * NDK;
  const float* lvp = log_var + (size_t)hm * NC * NDK;
  const float* lpp = log_prior + (size_t)hm * NC;

  float lp[NC];
  float lpm = -1e30f;
  for (int c = 0; c < NC; ++c) { lp[c] = lpp[c]; lpm = fmaxf(lpm, lp[c]); }
  float ls = 0.f;
  for (int c = 0; c < NC; ++c) ls += __expf(lp[c] - lpm);
  float lpn = lpm + __logf(ls);
  for (int c = 0; c < NC; ++c) lp[c] -= lpn;

  float a = alv[((size_t)b * NH + h) * NL + l];
  float avar = __expf(a);

  float lpdf[NC], msesum[NC], ivsum[NC], lvsum[NC];
  for (int c = 0; c < NC; ++c) {
    float ms = 0.f, iv = 0.f, lv2 = 0.f;
    const float* mc = mup + c * NDK;
    const float* lc = lvp + c * NDK;
    for (int d = 0; d < NDK; ++d) {
      float lvd = lc[d];
      float ivd = __expf(-lvd);
      float diff = z[d] - mc[d];
      ms += diff * diff * ivd;
      iv += ivd;
      lv2 += lvd;
    }
    msesum[c] = ms; ivsum[c] = iv; lvsum[c] = lv2;
    lpdf[c] = -0.5f * ms - 0.5f * lv2 - (float)NDK * PI_F + lp[c];
  }
  float m = fmaxf(fmaxf(lpdf[0], lpdf[1]), fmaxf(lpdf[2], lpdf[3]));
  float s = 0.f, e[NC];
  for (int c = 0; c < NC; ++c) { e[c] = __expf(lpdf[c] - m); s += e[c]; }
  float lses = m + __logf(s);

  float kl1 = 0.f, kl2 = 0.f;
  for (int c = 0; c < NC; ++c) {
    float logpr = lpdf[c] - lses;
    float cpc = e[c] / s;
    cp[(size_t)idx * NC + c] = cpc;
    lpdf_out[(size_t)idx * NC + c] = lpdf[c];
    kl1 += __expf(lp[c]) * (lp[c] - logpr);
    kl2 += 0.5f * cpc * (msesum[c] + avar * ivsum[c] + lvsum[c]);
  }
  float kl3 = -0.5f * (1.f + a) * (float)NDK;
  atomicAdd(&kl_out[b], (kl1 + kl2 + kl3) * (1.f / (float)(NH * NL)));

  if (doMax) {
    unsigned bits = __float_as_uint(m);
    unsigned enc = (m >= 0.f) ? (bits | 0x80000000u) : ~bits;
    atomicMax(gmax_enc, enc);
  }
}

// ===================== div_loss: tiles of cp.cp^T ==========================
__global__ __launch_bounds__(128) void dm_div_kernel(
    const float* __restrict__ cp, float* __restrict__ div_out)
{
  int bid = blockIdx.x;
  int qt = bid & 31, h = (bid >> 5) & 15, b = bid >> 9;
  int q0 = qt * 16;
  int tid = threadIdx.x, lane = tid & 31, wv = tid >> 5;
  int r = lane & 15, hlf = lane >> 4;
  const float* cph = cp + (((size_t)b * NH + h) * NL) * NC;
  float lsum = 0.f;
#if ENABLE_WMMA_X4 && __has_builtin(__builtin_amdgcn_wmma_f32_16x16x4_f32)
  v2f afr;
  afr[0] = cph[(q0 + r) * NC + 2 * hlf + 0];
  afr[1] = cph[(q0 + r) * NC + 2 * hlf + 1];
  for (int kt = wv; kt < 32; kt += 4) {
    int k0 = kt * 16;
    v2f bfr;
    bfr[0] = cph[(k0 + r) * NC + 2 * hlf + 0];
    bfr[1] = cph[(k0 + r) * NC + 2 * hlf + 1];
    v8f acc = {};
    acc = __builtin_amdgcn_wmma_f32_16x16x4_f32(false, afr, false, bfr,
                                                (short)0, acc, false, false);
#pragma unroll
    for (int i = 0; i < 8; ++i) {
      int q = q0 + hlf * 8 + i, k = k0 + r;
      float dlt = (q == k) ? 1.f : 0.f;
      float wgt = dlt * 1.25f + (1.f - dlt) * 0.75f;
      float d = acc[i] - dlt;
      lsum += d * d * wgt;
    }
  }
#else
  int q = q0 + (tid >> 3);
  float a0 = cph[q * NC + 0], a1 = cph[q * NC + 1];
  float a2 = cph[q * NC + 2], a3 = cph[q * NC + 3];
  for (int k = (tid & 7); k < NL; k += 8) {
    float g = a0 * cph[k * NC + 0] + a1 * cph[k * NC + 1] +
              a2 * cph[k * NC + 2] + a3 * cph[k * NC + 3];
    float dlt = (q == k) ? 1.f : 0.f;
    float wgt = dlt * 1.25f + (1.f - dlt) * 0.75f;
    float d = g - dlt;
    lsum += d * d * wgt;
  }
#endif
  __shared__ float red[128];
  red[tid] = lsum;
  __syncthreads();
  for (int s = 64; s > 0; s >>= 1) {
    if (tid < s) red[tid] += red[tid + s];
    __syncthreads();
  }
  if (tid == 0)
    atomicAdd(&div_out[b], red[0] * (1.f / ((float)NH * NL * NL)));
}

// ======================= fused attention kernel ============================
// Block per (b,h,qtile16); 128 threads = 4 waves.
// TDM loads K/V 64x64 fp32 tiles into shared KV[64][65] (async-tensor path);
// WMMA bf16 for scores and context; softmax+mask+renorm in between.
__global__ __launch_bounds__(128) void dm_attention_kernel(
    const float* __restrict__ Q, const float* __restrict__ Kmat,
    const float* __restrict__ V, const float* __restrict__ cpq,
    float* __restrict__ ctx_f, float* __restrict__ attn_out)
{
  int bid = blockIdx.x;
  int qt = bid & 31, h = (bid >> 5) & 15, b = bid >> 9;
  int q0 = qt * 16;
  int tid = threadIdx.x, lane = tid & 31, wv = tid >> 5;
  int r = lane & 15, hlf = lane >> 4;

  __shared__ __bf16 Qs[16][64 + 4];
  __shared__ float  KV[64][65];        // TDM dest, row pitch 65 dwords
  __shared__ float  S[16][512 + 4];
  __shared__ float  cpk[512][4];
  __shared__ float  cpq_s[16][4];
  __shared__ float  red[16][8];

  const unsigned kvLds = (unsigned)(unsigned long long)&KV[0][0];

  for (int i = tid; i < 16 * 64; i += 128) {
    int qq = i >> 6, dd = i & 63;
    Qs[qq][dd] = (__bf16)Q[((size_t)b * NL + q0 + qq) * ND + h * NDK + dd];
  }
  for (int i = tid; i < NL * NC; i += 128)
    cpk[i >> 2][i & 3] = cpq[(((size_t)b * NH + h) * NL) * NC + i];
  if (tid < 64)
    cpq_s[tid >> 2][tid & 3] =
        cpq[(((size_t)b * NH + h) * NL + q0) * NC + tid];
  __syncthreads();

  const float scale = 0.125f;  // 1/sqrt(64)
  for (int ko = 0; ko < NL; ko += 64) {
#if ENABLE_TDM
    if (wv == 0) {
      dm_tdm_load_2d(kvLds, Kmat + ((size_t)b * NL + ko) * ND + h * NDK,
                     64, 64, ND);
      __builtin_amdgcn_s_wait_tensorcnt(0);
    }
#else
    for (int i = tid; i < 64 * 64; i += 128) {
      int kk = i >> 6, dd = i & 63;
      KV[kk][dd] = Kmat[((size_t)b * NL + ko + kk) * ND + h * NDK + dd];
    }
#endif
    __syncthreads();
    v8f acc = {};
#pragma unroll
    for (int ks = 0; ks < 64; ks += 32) {
      v16bf af, bf;
#pragma unroll
      for (int j = 0; j < 16; ++j) {
        int km = ((j >> 3) << 4) + (hlf << 3) + (j & 7);
        af[j] = Qs[r][ks + km];
        bf[j] = (__bf16)KV[wv * 16 + r][ks + km];  // key-major tile
      }
      acc = __builtin_amdgcn_wmma_f32_16x16x32_bf16(false, af, false, bf,
                                                    (short)0, acc, false, false);
    }
#pragma unroll
    for (int i = 0; i < 8; ++i)
      S[hlf * 8 + i][ko + wv * 16 + r] = acc[i] * scale;
    __syncthreads();
  }

  // softmax + mask + renorm: 8 threads per row
  {
    int row = tid >> 3, seg = tid & 7;
    float mx = -1e30f;
    for (int kk = seg * 64; kk < seg * 64 + 64; ++kk)
      mx = fmaxf(mx, S[row][kk]);
    red[row][seg] = mx;
    __syncthreads();
    float rmax = red[row][0];
    for (int t = 1; t < 8; ++t) rmax = fmaxf(rmax, red[row][t]);
    __syncthreads();
    float sum = 0.f;
    for (int kk = seg * 64; kk < seg * 64 + 64; ++kk) {
      float e = __expf(S[row][kk] - rmax);
      S[row][kk] = e; sum += e;
    }
    red[row][seg] = sum;
    __syncthreads();
    float rsum = 0.f;
    for (int t = 0; t < 8; ++t) rsum += red[row][t];
    __syncthreads();
    float inv = 1.f / rsum;
    float msum = 0.f;
    for (int kk = seg * 64; kk < seg * 64 + 64; ++kk) {
      float g = cpq_s[row][0] * cpk[kk][0] + cpq_s[row][1] * cpk[kk][1] +
                cpq_s[row][2] * cpk[kk][2] + cpq_s[row][3] * cpk[kk][3];
      float av = S[row][kk] * inv * g;
      S[row][kk] = av; msum += av;
    }
    red[row][seg] = msum;
    __syncthreads();
    float rmsum = 0.f;
    for (int t = 0; t < 8; ++t) rmsum += red[row][t];
    float inv2 = 1.f / (rmsum + 1e-6f);
    for (int kk = seg * 64; kk < seg * 64 + 64; ++kk) {
      float av = S[row][kk] * inv2;
      S[row][kk] = av;
      atomicAdd(&attn_out[((size_t)b * NL + q0 + row) * NL + kk],
                av * (1.f / (float)NH));
    }
  }
  __syncthreads();

  // context: ctx[16x64] = P(16x512) * V(512x64); wave wv owns d-tile wv*16
  v8f cacc = {};
  for (int ko = 0; ko < NL; ko += 64) {
#if ENABLE_TDM
    if (wv == 0) {
      dm_tdm_load_2d(kvLds, V + ((size_t)b * NL + ko) * ND + h * NDK,
                     64, 64, ND);
      __builtin_amdgcn_s_wait_tensorcnt(0);
    }
#else
    for (int i = tid; i < 64 * 64; i += 128) {
      int kk = i >> 6, dd = i & 63;
      KV[kk][dd] = V[((size_t)b * NL + ko + kk) * ND + h * NDK + dd];
    }
#endif
    __syncthreads();
#pragma unroll
    for (int ks = 0; ks < 64; ks += 32) {
      v16bf af, bf;
#pragma unroll
      for (int j = 0; j < 16; ++j) {
        int km = ((j >> 3) << 4) + (hlf << 3) + (j & 7);
        af[j] = (__bf16)S[r][ko + ks + km];
        bf[j] = (__bf16)KV[ks + km][wv * 16 + r];
      }
      cacc = __builtin_amdgcn_wmma_f32_16x16x32_bf16(false, af, false, bf,
                                                     (short)0, cacc, false, false);
    }
    __syncthreads();
  }
#pragma unroll
  for (int i = 0; i < 8; ++i)
    ctx_f[((size_t)b * NL + q0 + hlf * 8 + i) * ND + h * NDK + wv * 16 + r] =
        cacc[i];
}

// =================== pdf + per-(b,h,c) weight sums (mi) ====================
__global__ __launch_bounds__(256) void dm_pdf_kernel(
    const float* __restrict__ lpdf, const float* __restrict__ cp,
    const unsigned int* __restrict__ gmax_enc,
    float* __restrict__ pdf, float* __restrict__ wsum)
{
  int idx = blockIdx.x * 256 + threadIdx.x;
  if (idx >= NB * NH * NL) return;
  unsigned enc = *gmax_enc;
  float gmax = (enc & 0x80000000u) ? __uint_as_float(enc & 0x7FFFFFFFu)
                                   : __uint_as_float(~enc);
  float s = 0.f;
  for (int c = 0; c < NC; ++c) s += __expf(lpdf[(size_t)idx * NC + c] - gmax);
  pdf[idx] = s;
  int h = (idx >> 9) & (NH - 1);
  int b = idx >> 13;
  for (int c = 0; c < NC; ++c)
    atomicAdd(&wsum[((size_t)b * NH + h) * NC + c],
              cp[(size_t)idx * NC + c] * s);
}

// ============== mi pairwise head term: per (b,l), 16x16 pairs ==============
__global__ __launch_bounds__(256) void dm_mi_pair_kernel(
    const float* __restrict__ cp_h, const float* __restrict__ pdf,
    const float* __restrict__ wsum, float* __restrict__ mi_out)
{
  int bl = blockIdx.x;
  int b = bl >> 9, l = bl & (NL - 1);
  __shared__ float mp[NH][NC], cpl[NH][NC];
  int tid = threadIdx.x;
  if (tid < NH * NC) {
    int i = tid >> 2, c = tid & 3;
    size_t idx = ((size_t)b * NH + i) * NL + l;
    float wvl = cp_h[idx * NC + c] * pdf[idx];
    mp[i][c] = wvl / fmaxf(wsum[((size_t)b * NH + i) * NC + c], 1e-6f);
    cpl[i][c] = cp_h[idx * NC + c];
  }
  __syncthreads();
  float acc = 0.f;
  {
    int i = tid >> 4, j = tid & 15;
    if (i != j) {
      float p = mp[i][0] * cpl[j][0] + mp[i][1] * cpl[j][1] +
                mp[i][2] * cpl[j][2] + mp[i][3] * cpl[j][3];
      p = fminf(p, 1.f);
      acc = -__logf(1.f - p + 1e-6f);
    }
  }
  __shared__ float red[256];
  red[tid] = acc;
  __syncthreads();
  for (int s = 128; s > 0; s >>= 1) {
    if (tid < s) red[tid] += red[tid + s];
    __syncthreads();
  }
  if (tid == 0)
    atomicAdd(&mi_out[b], red[0] * (10.f / ((float)NL * NH * NH)));
}

// ======= mi diag term: diag(log_softmax(cp.cp^T)) per (b,h,q) row ==========
__global__ __launch_bounds__(256) void dm_mi_diag_kernel(
    const float* __restrict__ cp_h, float* __restrict__ mi_out)
{
  int idx = blockIdx.x * 256 + threadIdx.x;
  if (idx >= NB * NH * NL) return;
  int q = idx & (NL - 1);
  int h = (idx >> 9) & (NH - 1);
  int b = idx >> 13;
  const float* cph = cp_h + (((size_t)b * NH + h) * NL) * NC;
  float a0 = cph[q * NC + 0], a1 = cph[q * NC + 1];
  float a2 = cph[q * NC + 2], a3 = cph[q * NC + 3];
  float m = -1e30f, s = 0.f, diag = 0.f;
  for (int k = 0; k < NL; ++k) {
    float v = a0 * cph[k * NC + 0] + a1 * cph[k * NC + 1] +
              a2 * cph[k * NC + 2] + a3 * cph[k * NC + 3];
    if (k == q) diag = v;
    if (v > m) { s = s * __expf(m - v) + 1.f; m = v; }
    else s += __expf(v - m);
  }
  float lse = m + __logf(s);
  atomicAdd(&mi_out[b], -(diag - lse) / ((float)NH * NL));
}

// ================================ launch ===================================
extern "C" void kernel_launch(void* const* d_in, const int* in_sizes, int n_in,
                              void* d_out, int out_size, void* d_ws, size_t ws_size,
                              hipStream_t stream) {
  const float* query    = (const float*)d_in[0];
  const float* key      = (const float*)d_in[1];
  const float* value    = (const float*)d_in[2];
  const float* Wq       = (const float*)d_in[3];
  const float* bq       = (const float*)d_in[4];
  const float* Wk       = (const float*)d_in[5];
  const float* bk       = (const float*)d_in[6];
  const float* Wv       = (const float*)d_in[7];
  const float* bv       = (const float*)d_in[8];
  const float* Wo       = (const float*)d_in[9];
  const float* bo       = (const float*)d_in[10];
  const float* Wsem     = (const float*)d_in[11];
  const float* bsem     = (const float*)d_in[12];
  const float* Whead    = (const float*)d_in[13];
  const float* bhead    = (const float*)d_in[14];
  const float* tok_mu   = (const float*)d_in[15];
  const float* tok_lv   = (const float*)d_in[16];
  const float* tok_lp   = (const float*)d_in[17];
  const float* head_mu  = (const float*)d_in[18];
  const float* head_lv  = (const float*)d_in[19];
  const float* head_lp  = (const float*)d_in[20];

  const size_t BLD = (size_t)NB * NL * ND;      // 4,194,304
  const size_t BHL = (size_t)NB * NH * NL;      //    65,536
  const size_t BHLC = BHL * NC;                 //   262,144

  float* ws = (float*)d_ws;
  float* qb     = ws;            ws += BLD;
  float* kb     = ws;            ws += BLD;
  float* vb     = ws;            ws += BLD;
  float* ctx_f  = ws;            ws += BLD;
  float* alv    = ws;            ws += BHL;
  float* alv_h  = ws;            ws += BHL;
  float* cp_q   = ws;            ws += BHLC;
  float* cp_h   = ws;            ws += BHLC;
  float* lpdf   = ws;            ws += BHLC;
  float* pdf    = ws;            ws += BHL;
  float* wsum   = ws;            ws += NB * NH * NC;   // 512
  unsigned int* gmax = (unsigned int*)ws;              // 1 (zero == -inf enc)

  float* out_p    = (float*)d_out;                 // (B,L,D)
  float* attn_p   = out_p + BLD;                   // (B,L,L)
  float* kl_p     = attn_p + (size_t)NB * NL * NL; // (B)
  float* div_p    = kl_p + NB;
  float* mi_p     = div_p + NB;

  {
    int n1 = NB * NL * NL + 3 * NB;
    dm_zero_kernel<<<(n1 + 255) / 256, 256, 0, stream>>>(attn_p, n1);
    int n2 = NB * NH * NC + 1;
    dm_zero_kernel<<<(n2 + 255) / 256, 256, 0, stream>>>(wsum, n2);
  }

  dim3 ggrid(32, 16);  // M/128, N/64
  dm_gemm_bias_kernel<<<ggrid, 256, 0, stream>>>(query, Wq, bq, qb, 4096, 1024, 1024);
  dm_gemm_bias_kernel<<<ggrid, 256, 0, stream>>>(key,   Wk, bk, kb, 4096, 1024, 1024);
  dm_gemm_bias_kernel<<<ggrid, 256, 0, stream>>>(value, Wv, bv, vb, 4096, 1024, 1024);
  dm_proj16_kernel<<<NB * NL, 64, 0, stream>>>(qb, Wsem, bsem, alv);
  dm_clustering_kernel<<<(int)((BHL + 255) / 256), 256, 0, stream>>>(
      qb, alv, tok_mu, tok_lv, tok_lp, cp_q, lpdf, kl_p, gmax, /*muHStride=*/1,
      /*doMax=*/0);
  dm_div_kernel<<<NB * NH * 32, 128, 0, stream>>>(cp_q, div_p);
  dm_attention_kernel<<<NB * NH * 32, 128, 0, stream>>>(qb, kb, vb, cp_q,
                                                        ctx_f, attn_p);
  dm_gemm_bias_kernel<<<ggrid, 256, 0, stream>>>(ctx_f, Wo, bo, out_p, 4096, 1024, 1024);
  dm_proj16_kernel<<<NB * NL, 64, 0, stream>>>(ctx_f, Whead, bhead, alv_h);
  dm_clustering_kernel<<<(int)((BHL + 255) / 256), 256, 0, stream>>>(
      ctx_f, alv_h, head_mu, head_lv, head_lp, cp_h, lpdf, kl_p, gmax,
      /*muHStride=*/0, /*doMax=*/1);
  dm_div_kernel<<<NB * NH * 32, 128, 0, stream>>>(cp_h, div_p);
  dm_pdf_kernel<<<(int)((BHL + 255) / 256), 256, 0, stream>>>(lpdf, cp_h, gmax,
                                                              pdf, wsum);
  dm_mi_pair_kernel<<<NB * NL, 256, 0, stream>>>(cp_h, pdf, wsum, mi_p);
  dm_mi_diag_kernel<<<(int)((BHL + 255) / 256), 256, 0, stream>>>(cp_h, mi_p);

  (void)in_sizes; (void)n_in; (void)out_size; (void)ws_size;
}
